// GNN_26414048871137
// MI455X (gfx1250) — matrix operand order repaired and checked by reading, not compiled, not run on hardware
//
#include <hip/hip_runtime.h>
#include <hip/hip_fp16.h>

// ---------------- constants (match reference) ----------------
#define HEADS      2
#define NEG_SLOPE  0.2f
#define PN_EPS     1e-5f

typedef __attribute__((ext_vector_type(16))) _Float16 v16h;
typedef __attribute__((ext_vector_type(8)))  float    v8f;

// Order-preserving float -> unsigned encoding for atomicMax-based segment_max.
__device__ __forceinline__ unsigned encF(float f) {
    unsigned u = __float_as_uint(f);
    return (u & 0x80000000u) ? ~u : (u | 0x80000000u);
}
__device__ __forceinline__ float decF(unsigned u) {
    u = (u & 0x80000000u) ? (u & 0x7FFFFFFFu) : ~u;
    return __uint_as_float(u);
}
// encF(-inf) == ~0xFF800000 == 0x007FFFFF
#define ENC_NEG_INF 0x007FFFFFu

// ---------------- WMMA GEMM: XH[M,Fout] = X[M,128] * W[128,Fout] ----------------
// One wave per 16x16 output tile; K=128 fully unrolled into 4 chained
// v_wmma_f32_16x16x32_f16 ops (f16 operands, f32 accumulate).
// Fragment layouts per CDNA5 ISA 7.12.2 (wave32):
//   A: lane L -> row L%16; hi=L/16; halves cover K = kb+hi*8..+7 and kb+16+hi*8..+7.
//   B: lane L -> col L%16; hi=L/16; halves cover K = kb+hi*8..+7 and kb+16+hi*8..+7.
//   D: row = tileM*16 + vgpr + 8*(L/16); col = tileN*16 + L%16.
__global__ void gat_gemm_wmma_kernel(const float* __restrict__ X,
                                     const float* __restrict__ W,
                                     float* __restrict__ XH,
                                     int M, int Fout) {
    const int K = 128;
    int wave   = (blockIdx.x * blockDim.x + threadIdx.x) >> 5;
    int lane   = threadIdx.x & 31;
    int tilesN = Fout >> 4;
    int tileM  = wave / tilesN;
    int tileN  = wave % tilesN;
    if (tileM * 16 >= M) return;

    int lo = lane & 15;
    int hi = lane >> 4;
    const float* xrow = X + (size_t)(tileM * 16 + lo) * K;
    int bcol = tileN * 16 + lo;

    v8f acc = {};
#pragma unroll
    for (int kb = 0; kb < K; kb += 32) {
        // A fragment: two contiguous 8-float runs -> 4x global_load_b128
        const float4 a0 = *(const float4*)(xrow + kb + hi * 8);
        const float4 a1 = *(const float4*)(xrow + kb + hi * 8 + 4);
        const float4 a2 = *(const float4*)(xrow + kb + 16 + hi * 8);
        const float4 a3 = *(const float4*)(xrow + kb + 16 + hi * 8 + 4);
        v16h a;
        a[0]  = (_Float16)a0.x;  a[1]  = (_Float16)a0.y;
        a[2]  = (_Float16)a0.z;  a[3]  = (_Float16)a0.w;
        a[4]  = (_Float16)a1.x;  a[5]  = (_Float16)a1.y;
        a[6]  = (_Float16)a1.z;  a[7]  = (_Float16)a1.w;
        a[8]  = (_Float16)a2.x;  a[9]  = (_Float16)a2.y;
        a[10] = (_Float16)a2.z;  a[11] = (_Float16)a2.w;
        a[12] = (_Float16)a3.x;  a[13] = (_Float16)a3.y;
        a[14] = (_Float16)a3.z;  a[15] = (_Float16)a3.w;
        // B fragment: column-strided (stride Fout); W is 64KB, L2-resident
        const float* wp = W + (size_t)(kb + hi * 8) * Fout + bcol;
        v16h b;
#pragma unroll
        for (int t = 0; t < 16; ++t) {
            b[t] = (_Float16)wp[(size_t)(t + (t >= 8 ? 8 : 0)) * Fout];
        }
        acc = __builtin_amdgcn_wmma_f32_16x16x32_f16(
            /*neg_a=*/false, a, /*neg_b=*/false, b,
            /*c_mod=*/(short)0, acc, /*reuse_a=*/false, /*reuse_b=*/false);
    }

    int row0 = tileM * 16 + hi * 8;
#pragma unroll
    for (int r = 0; r < 8; ++r) {
        XH[(size_t)(row0 + r) * Fout + bcol] = acc[r];
    }
}

// ---------------- attention coefficients: al[n,h] = xh[n,h,:] . a[h,:] ----------------
__global__ void attn_coef_kernel(const float* __restrict__ XH,
                                 const float* __restrict__ a_src,
                                 const float* __restrict__ a_dst,
                                 float* __restrict__ alS, float* __restrict__ alD,
                                 int N, int D) {
    int idx = blockIdx.x * blockDim.x + threadIdx.x;  // n*HEADS + h
    if (idx >= N * HEADS) return;
    int h = idx % HEADS, n = idx / HEADS;
    const float* row = XH + (size_t)n * HEADS * D + h * D;
    const float* as = a_src + h * D;
    const float* ad = a_dst + h * D;
    float s = 0.f, d = 0.f;
    for (int i = 0; i < D; ++i) {
        float v = row[i];
        s += v * as[i];
        d += v * ad[i];
    }
    alS[idx] = s;
    alD[idx] = d;
}

// ---------------- generic 32-bit fill ----------------
__global__ void fill_u32_kernel(unsigned* __restrict__ p, unsigned v, long long n) {
    long long i = blockIdx.x * (long long)blockDim.x + threadIdx.x;
    if (i < n) p[i] = v;
}

// ---------------- per-edge segment max (softmax pass 1) ----------------
__global__ void edge_max_kernel(const int* __restrict__ src, const int* __restrict__ dst,
                                const float* __restrict__ alS, const float* __restrict__ alD,
                                unsigned* __restrict__ m_enc, int E, int N) {
    int e = blockIdx.x * blockDim.x + threadIdx.x;
    if (e >= E + N) return;
    int s = (e < E) ? src[e] : (e - E);   // self-loops appended
    int d = (e < E) ? dst[e] : (e - E);
#pragma unroll
    for (int h = 0; h < HEADS; ++h) {
        float l = alS[s * HEADS + h] + alD[d * HEADS + h];
        l = (l > 0.f) ? l : NEG_SLOPE * l;
        atomicMax(&m_enc[d * HEADS + h], encF(l));
    }
}

// ---------------- per-edge exp + segment sum (softmax pass 2) ----------------
__global__ void edge_exp_kernel(const int* __restrict__ src, const int* __restrict__ dst,
                                const float* __restrict__ alS, const float* __restrict__ alD,
                                const unsigned* __restrict__ m_enc,
                                float* __restrict__ eArr, float* __restrict__ sArr,
                                int E, int N) {
    int e = blockIdx.x * blockDim.x + threadIdx.x;
    if (e >= E + N) return;
    int s = (e < E) ? src[e] : (e - E);
    int d = (e < E) ? dst[e] : (e - E);
#pragma unroll
    for (int h = 0; h < HEADS; ++h) {
        float l = alS[s * HEADS + h] + alD[d * HEADS + h];
        l = (l > 0.f) ? l : NEG_SLOPE * l;
        float m  = decF(m_enc[d * HEADS + h]);
        float ev = expf(l - m);
        eArr[(size_t)e * HEADS + h] = ev;
        atomicAdd(&sArr[d * HEADS + h], ev);
    }
}

// ---------------- message aggregation: AGG[dst] += alpha * XH[src] ----------------
// One thread per (edge, 4-feature chunk); float4 gather + 4 L2 atomic f32 adds.
__global__ void edge_agg_kernel(const int* __restrict__ src, const int* __restrict__ dst,
                                const float* __restrict__ eArr, const float* __restrict__ sArr,
                                const float* __restrict__ XH, float* __restrict__ AGG,
                                int E, int N, int D) {
    const int HD = HEADS * D;
    const int chunks = HD / 4;
    long long idx = blockIdx.x * (long long)blockDim.x + threadIdx.x;
    long long e = idx / chunks;
    if (e >= (long long)(E + N)) return;
    int c  = (int)(idx % chunks);
    int s  = (e < E) ? src[e] : (int)(e - E);
    int d  = (e < E) ? dst[e] : (int)(e - E);
    int hd = c * 4;
    int h  = hd / D;
    float alpha = eArr[(size_t)e * HEADS + h] / sArr[d * HEADS + h];
    const float4 x4 = *(const float4*)(XH + (size_t)s * HD + hd);
    float* o = AGG + (size_t)d * HD + hd;
    atomicAdd(o + 0, alpha * x4.x);
    atomicAdd(o + 1, alpha * x4.y);
    atomicAdd(o + 2, alpha * x4.z);
    atomicAdd(o + 3, alpha * x4.w);
}

// ---------------- column sums for PairNorm centering ----------------
__global__ void colsum_kernel(const float* __restrict__ X, float* __restrict__ colsum,
                              int N, int C, int rowsPerBlock) {
    int c  = threadIdx.x;
    int r0 = blockIdx.x * rowsPerBlock;
    int r1 = r0 + rowsPerBlock;
    if (r1 > N) r1 = N;
    float acc = 0.f;
    for (int r = r0; r < r1; ++r) acc += X[(size_t)r * C + c];
    atomicAdd(&colsum[c], acc);
}

// ---------------- PairNorm(scale_individually) + ReLU, in place ----------------
// One wave32 per node row of C=128; 4 features/lane; shuffle reduction for ||y||.
__global__ void pairnorm_relu_kernel(float* __restrict__ X, const float* __restrict__ colsum,
                                     int N, int C /*==128*/) {
    int wave = (blockIdx.x * blockDim.x + threadIdx.x) >> 5;
    int lane = threadIdx.x & 31;
    if (wave >= N) return;
    float* row = X + (size_t)wave * C;
    float invN = 1.0f / (float)N;
    float v[4];
    float ss = 0.f;
#pragma unroll
    for (int i = 0; i < 4; ++i) {
        int c = lane * 4 + i;
        float y = row[c] - colsum[c] * invN;
        v[i] = y;
        ss += y * y;
    }
#pragma unroll
    for (int off = 16; off > 0; off >>= 1) ss += __shfl_xor(ss, off, 32);
    float scale = 1.0f / (PN_EPS + sqrtf(ss));
#pragma unroll
    for (int i = 0; i < 4; ++i) {
        float o = v[i] * scale;
        row[lane * 4 + i] = (o > 0.f) ? o : 0.f;
    }
}

// ---------------- layer-3 epilogue: mean over heads + bias ----------------
__global__ void head_mean_kernel(const float* __restrict__ AGG, const float* __restrict__ b,
                                 float* __restrict__ OUT, int N, int D) {
    int idx = blockIdx.x * blockDim.x + threadIdx.x;
    if (idx >= N * D) return;
    int n = idx / D, c = idx % D;
    float acc = 0.f;
#pragma unroll
    for (int h = 0; h < HEADS; ++h) acc += AGG[(size_t)n * HEADS * D + h * D + c];
    OUT[idx] = acc * (1.0f / HEADS) + b[c];
}

// =========================== host orchestration ===========================
extern "C" void kernel_launch(void* const* d_in, const int* in_sizes, int n_in,
                              void* d_out, int out_size, void* d_ws, size_t ws_size,
                              hipStream_t stream) {
    const float* x   = (const float*)d_in[0];
    const int*   ei  = (const int*)d_in[1];
    const float* W1  = (const float*)d_in[2];
    const float* as1 = (const float*)d_in[3];
    const float* ad1 = (const float*)d_in[4];
    // d_in[5] = b1 (zeros; cancels under PairNorm centering)
    const float* W2  = (const float*)d_in[6];
    const float* as2 = (const float*)d_in[7];
    const float* ad2 = (const float*)d_in[8];
    // d_in[9] = b2 (same)
    const float* W3  = (const float*)d_in[10];
    const float* as3 = (const float*)d_in[11];
    const float* ad3 = (const float*)d_in[12];
    const float* b3  = (const float*)d_in[13];
    float* out = (float*)d_out;

    const int N    = in_sizes[0] / 128;   // 50000
    const int E    = in_sizes[1] / 2;     // 800000
    const int Etot = E + N;               // self-loops appended
    const int* srcIdx = ei;               // edge_index[0]
    const int* dstIdx = ei + E;           // edge_index[1]

    // ---- workspace carve-up (floats) ----
    float* ws = (float*)d_ws;
    size_t NN = (size_t)N;
    float*    BUF1   = ws;                         // [N,128]  xh (per layer)
    float*    BUF2   = BUF1 + NN * 128;            // [N,128]  agg / h
    float*    BUF3   = BUF2 + NN * 128;            // [N,128]  agg / h
    float*    alS    = BUF3 + NN * 128;            // [N,2]
    float*    alD    = alS + NN * HEADS;           // [N,2]
    unsigned* m_enc  = (unsigned*)(alD + NN * HEADS);      // [N,2]
    float*    sArr   = (float*)m_enc + NN * HEADS;         // [N,2]
    float*    eArr   = sArr + NN * HEADS;          // [E+N,2]
    float*    colsum = eArr + (size_t)Etot * HEADS;        // [128]
    (void)ws_size; (void)n_in; (void)out_size;

    auto cdiv = [](long long a, long long b) { return (int)((a + b - 1) / b); };

    auto run_layer = [&](const float* Xin, const float* W, const float* a_s,
                         const float* a_d, float* XH, float* AGG, int D) {
        const int Fout = HEADS * D;
        // 1) feature transform via WMMA
        int waves = (N / 16) * (Fout / 16);
        gat_gemm_wmma_kernel<<<cdiv((long long)waves * 32, 256), 256, 0, stream>>>(
            Xin, W, XH, N, Fout);
        // 2) per-node attention coefficients
        attn_coef_kernel<<<cdiv((long long)N * HEADS, 256), 256, 0, stream>>>(
            XH, a_s, a_d, alS, alD, N, D);
        // 3) init scratch (deterministic every call)
        fill_u32_kernel<<<cdiv(NN * HEADS, 256), 256, 0, stream>>>(m_enc, ENC_NEG_INF, NN * HEADS);
        fill_u32_kernel<<<cdiv(NN * HEADS, 256), 256, 0, stream>>>((unsigned*)sArr, 0u, NN * HEADS);
        fill_u32_kernel<<<cdiv(NN * Fout, 256), 256, 0, stream>>>((unsigned*)AGG, 0u, NN * Fout);
        // 4) softmax passes over edges
        edge_max_kernel<<<cdiv(Etot, 256), 256, 0, stream>>>(srcIdx, dstIdx, alS, alD, m_enc, E, N);
        edge_exp_kernel<<<cdiv(Etot, 256), 256, 0, stream>>>(srcIdx, dstIdx, alS, alD, m_enc,
                                                             eArr, sArr, E, N);
        // 5) weighted aggregation (dominant cost: gather + L2 atomic adds)
        long long aggThreads = (long long)Etot * (Fout / 4);
        edge_agg_kernel<<<cdiv(aggThreads, 256), 256, 0, stream>>>(srcIdx, dstIdx, eArr, sArr,
                                                                   XH, AGG, E, N, D);
    };

    auto pairnorm_relu = [&](float* H) {
        fill_u32_kernel<<<1, 128, 0, stream>>>((unsigned*)colsum, 0u, 128);
        colsum_kernel<<<cdiv(N, 256), 128, 0, stream>>>(H, colsum, N, 128, 256);
        pairnorm_relu_kernel<<<cdiv((long long)N * 32, 256), 256, 0, stream>>>(H, colsum, N, 128);
    };

    // Layer 1: x -> BUF2
    run_layer(x, W1, as1, ad1, BUF1, BUF2, 64);
    pairnorm_relu(BUF2);
    // Layer 2: BUF2 -> BUF3
    run_layer(BUF2, W2, as2, ad2, BUF1, BUF3, 64);
    pairnorm_relu(BUF3);
    // Layer 3: BUF3 -> BUF2[:, :32], then head-mean + bias -> out
    run_layer(BUF3, W3, as3, ad3, BUF1, BUF2, 16);
    head_mean_kernel<<<cdiv((long long)N * 16, 256), 256, 0, stream>>>(BUF2, b3, out, N, 16);
}
